// EGNN_4174708212115
// MI455X (gfx1250) — matrix operand order repaired and checked by reading, not compiled, not run on hardware
//
#include <hip/hip_runtime.h>
#include <math.h>

// ---------------------------------------------------------------------------
// EGNN on gfx1250 (MI455X): all matmuls via v_wmma_f32_16x16x32_f16.
// Wave32 fragment layouts per CDNA5 ISA 7.12.2:
//   A (16x32 f16): lane<16 -> M=lane; VGPR v holds K pair
//       K = 2v + 16*(lane>=16) (this code's order, not the ISA table)
//   B (32x16 f16): lane%16 = N; VGPR v holds K pair K = 2v + 16*(lane>=16), same as A (code's choice)
//   C/D (16x16 f32): lane%16 = N; VGPR v -> M = v + 8*(lane>=16)
// Weights stored TRANSPOSED in LDS (WT[n][k]) so B pairs are contiguous.
// All staging tiles are per-wave; CDNA5 same-wave DS ops are in-order, so no
// block barriers are needed inside the tile loop (only after weight preload).
// Matmul operands (h, edge_feat, h_neigh) are pre-converted to f16 so the
// gather path is pure b128 copies; SiLU uses v_rcp_f32 instead of IEEE fdiv.
// ---------------------------------------------------------------------------

typedef _Float16 h16;
typedef __attribute__((ext_vector_type(16))) _Float16 v16h;
typedef __attribute__((ext_vector_type(8)))  _Float16 v8h;   // 16 bytes
typedef __attribute__((ext_vector_type(8)))  float    v8f;

__device__ __forceinline__ float rcp_f(float x) { return __builtin_amdgcn_rcpf(x); }
__device__ __forceinline__ float silu_f(float x) { return x * rcp_f(1.0f + __expf(-x)); }

__device__ __forceinline__ v16h load_a16(const h16* __restrict__ t, int stride) {
  const int lane = threadIdx.x & 31;
  const int hh = lane >> 4, m = lane & 15;
  v16h a;
#pragma unroll
  for (int v = 0; v < 8; ++v) {
    const int kb = 2 * v + 16 * hh;
    a[2 * v]     = t[m * stride + kb];
    a[2 * v + 1] = t[m * stride + kb + 1];
  }
  return a;
}

__device__ __forceinline__ v16h load_b16(const h16* __restrict__ wt, int strideK, int koff) {
  const int lane = threadIdx.x & 31;
  const int n = lane & 15;
  const h16* p = wt + n * strideK + koff + 16 * (lane >> 4);
  v16h b;
#pragma unroll
  for (int v = 0; v < 8; ++v) {
    b[2 * v]     = p[2 * v];
    b[2 * v + 1] = p[2 * v + 1];
  }
  return b;
}

__device__ __forceinline__ v8f wmma16(v16h a, v16h b, v8f c) {
  return __builtin_amdgcn_wmma_f32_16x16x32_f16(false, a, false, b, (short)0, c, false, false);
}

// ---------------------------------------------------------------------------
// Edge kernel. Stage-1 input layout (K padded 145->160, ef/radial permuted):
//   K   0..63 : h_src      64..127 : h_dst
//   K 128..143: edge_feat[0..15]   K 144: radial   K 145..159: zero
// sWT1 applies the same permutation to ew1^T, so results match the reference.
// ---------------------------------------------------------------------------
__global__ void __launch_bounds__(128) egnn_edge_kernel(
    const h16* __restrict__ h, const float* __restrict__ x,
    const h16* __restrict__ efh, const int* __restrict__ src, const int* __restrict__ dst,
    const float* __restrict__ ew1, const float* __restrict__ eb1,
    const float* __restrict__ ew2, const float* __restrict__ eb2,
    const float* __restrict__ cw1, const float* __restrict__ cb1,
    const float* __restrict__ cw2,
    float* __restrict__ h_neigh, float* __restrict__ x_sum, float* __restrict__ degv,
    int E) {
  __shared__ alignas(16) h16 sWT1[64 * 160];   // permuted ew1^T
  __shared__ alignas(16) h16 sWT2[64 * 64];    // ew2^T
  __shared__ alignas(16) h16 sWT3[64 * 64];    // cw1^T
  __shared__ float sCW2[64], sEB1[64], sEB2[64], sCB1[64];
  __shared__ alignas(16) h16 sF[4][16 * 32];   // per-wave A chunk
  __shared__ alignas(16) h16 sT1[4][16 * 64];  // per-wave t1 (later t3)
  __shared__ alignas(16) h16 sT2[4][16 * 64];  // per-wave mh
  __shared__ float sXdn[4][16 * 3];
  __shared__ float sRad[4][16];
  __shared__ int sSrcI[4][16];
  __shared__ int sDstI[4][16];

  const int tid = threadIdx.x;
  for (int i = tid; i < 64 * 160; i += 128) {
    const int n = i / 160, k = i - n * 160;
    float wv = 0.0f;
    if (k < 128)       wv = ew1[(size_t)k * 64 + n];          // h_src | h_dst
    else if (k < 144)  wv = ew1[(size_t)(k + 1) * 64 + n];    // ef[k-128] -> ref 129+(k-128)
    else if (k == 144) wv = ew1[(size_t)128 * 64 + n];        // radial
    sWT1[i] = (h16)wv;
  }
  for (int i = tid; i < 64 * 64; i += 128) {
    const int n = i >> 6, k = i & 63;
    sWT2[i] = (h16)ew2[k * 64 + n];
    sWT3[i] = (h16)cw1[k * 64 + n];
  }
  if (tid < 64) { sCW2[tid] = cw2[tid]; sEB1[tid] = eb1[tid]; sEB2[tid] = eb2[tid]; sCB1[tid] = cb1[tid]; }
  __syncthreads();   // only block-wide barrier: weights visible to all waves

  const int wave = tid >> 5, lane = tid & 31;
  const int n16 = lane & 15, hh = lane >> 4;
  h16* F = sF[wave]; h16* T1 = sT1[wave]; h16* T2 = sT2[wave];
  float* Xdn = sXdn[wave]; float* Rad = sRad[wave];
  int* Si = sSrcI[wave]; int* Di = sDstI[wave];

  const int waveGlobal = blockIdx.x * 4 + wave;
  const int totalWaves = gridDim.x * 4;
  const int nTiles = (E + 15) >> 4;

  for (int tile = waveGlobal; tile < nTiles; tile += totalWaves) {
    const int base = tile << 4;

    if (lane < 16) {
      const int e = base + lane;
      const bool ev = e < E;
      const int s = ev ? src[e] : -1;
      const int d = ev ? dst[e] : -1;
      Si[lane] = s; Di[lane] = d;
      float r = 0.f, nx = 0.f, ny = 0.f, nz = 0.f;
      if (ev) {
        const float dx = x[s * 3 + 0] - x[d * 3 + 0];
        const float dy = x[s * 3 + 1] - x[d * 3 + 1];
        const float dz = x[s * 3 + 2] - x[d * 3 + 2];
        r = dx * dx + dy * dy + dz * dz;
        const float inv = rcp_f(sqrtf(r) + 1e-30f);
        nx = dx * inv; ny = dy * inv; nz = dz * inv;
      }
      Rad[lane] = r;
      Xdn[lane * 3 + 0] = nx; Xdn[lane * 3 + 1] = ny; Xdn[lane * 3 + 2] = nz;
    }

    const int er = lane >> 1;           // edge row this lane stages
    const int coff = (lane & 1) * 16;   // 16-col half
    const v8h vz = {};

    // ---- Stage 1: 5 K-chunks x 4 N-tiles ----
    v8f acc[4] = {};
#pragma unroll
    for (int ck = 0; ck < 5; ++ck) {
      if (ck < 4) {
        const int gi = (ck < 2) ? Si[er] : Di[er];
        const bool ok = gi >= 0;
        const int kb = (ck & 1) * 32 + coff;
        const v8h* hp = (const v8h*)(h + (size_t)(ok ? gi : 0) * 64 + kb);
        const v8h q0 = hp[0], q1 = hp[1];
        *(v8h*)&F[er * 32 + coff]     = ok ? q0 : vz;
        *(v8h*)&F[er * 32 + coff + 8] = ok ? q1 : vz;
      } else if (coff == 0) {           // K 128..143 = edge_feat
        const int gd = Di[er];
        const bool ok = gd >= 0;
        const v8h* ep = (const v8h*)(efh + (size_t)(ok ? (base + er) : 0) * 16);
        const v8h q0 = ep[0], q1 = ep[1];
        *(v8h*)&F[er * 32]     = ok ? q0 : vz;
        *(v8h*)&F[er * 32 + 8] = ok ? q1 : vz;
      } else {                          // K 144 = radial, rest zero
        v8h p = {};
        p[0] = (h16)Rad[er];
        *(v8h*)&F[er * 32 + 16] = p;
        *(v8h*)&F[er * 32 + 24] = vz;
      }
      const v16h a = load_a16(F, 32);
#pragma unroll
      for (int nt = 0; nt < 4; ++nt)
        acc[nt] = wmma16(a, load_b16(sWT1 + nt * 16 * 160, 160, ck * 32), acc[nt]);
    }
#pragma unroll
    for (int nt = 0; nt < 4; ++nt) {
      const int col = nt * 16 + n16;
      const float b = sEB1[col];
#pragma unroll
      for (int v = 0; v < 8; ++v)
        T1[(v + 8 * hh) * 64 + col] = (h16)silu_f(acc[nt][v] + b);
    }

    // ---- Stage 2: mh = silu(t1 @ ew2 + eb2); scatter into h_neigh ----
    v8f m[4] = {};
#pragma unroll
    for (int ck = 0; ck < 2; ++ck) {
      const v16h a = load_a16(T1 + ck * 32, 64);
#pragma unroll
      for (int nt = 0; nt < 4; ++nt)
        m[nt] = wmma16(a, load_b16(sWT2 + nt * 16 * 64, 64, ck * 32), m[nt]);
    }
#pragma unroll
    for (int nt = 0; nt < 4; ++nt) {
      const int col = nt * 16 + n16;
      const float b = sEB2[col];
#pragma unroll
      for (int v = 0; v < 8; ++v) {
        const int row = v + 8 * hh;
        const float y = silu_f(m[nt][v] + b);
        T2[row * 64 + col] = (h16)y;
        const int d = Di[row];
        if (d >= 0) atomicAdd(&h_neigh[(size_t)d * 64 + col], y);
      }
    }

    // ---- Stage 3: t3 = silu(mh @ cw1 + cb1) (into T1) ----
    v8f t[4] = {};
#pragma unroll
    for (int ck = 0; ck < 2; ++ck) {
      const v16h a = load_a16(T2 + ck * 32, 64);
#pragma unroll
      for (int nt = 0; nt < 4; ++nt)
        t[nt] = wmma16(a, load_b16(sWT3 + nt * 16 * 64, 64, ck * 32), t[nt]);
    }
#pragma unroll
    for (int nt = 0; nt < 4; ++nt) {
      const int col = nt * 16 + n16;
      const float b = sCB1[col];
#pragma unroll
      for (int v = 0; v < 8; ++v)
        T1[(v + 8 * hh) * 64 + col] = (h16)silu_f(t[nt][v] + b);
    }

    // ---- Stage 4: coef = t3 . cw2 ; scatter coef*xdn and degree ----
    {
      const int r = lane & 15, ko = (lane >> 4) * 32;
      float acc2 = 0.0f;
#pragma unroll
      for (int k = 0; k < 32; ++k)
        acc2 += (float)T1[r * 64 + ko + k] * sCW2[ko + k];
      acc2 += __shfl_xor(acc2, 16, 32);   // combine the two half-rows
      const int d = Di[r];
      if (lane < 16 && d >= 0) {
        atomicAdd(&x_sum[(size_t)d * 3 + 0], acc2 * Xdn[r * 3 + 0]);
        atomicAdd(&x_sum[(size_t)d * 3 + 1], acc2 * Xdn[r * 3 + 1]);
        atomicAdd(&x_sum[(size_t)d * 3 + 2], acc2 * Xdn[r * 3 + 2]);
        atomicAdd(&degv[d], 1.0f);
      }
    }
  }
}

// ---------------------------------------------------------------------------
// Node kernel: hn = silu([h|h_neigh] @ nw1 + nb1) @ nw2 + nb2; BN partial sums.
// ---------------------------------------------------------------------------
__global__ void __launch_bounds__(128) egnn_node_kernel(
    const h16* __restrict__ h, const h16* __restrict__ h_neighH,
    const float* __restrict__ nw1, const float* __restrict__ nb1,
    const float* __restrict__ nw2, const float* __restrict__ nb2,
    float* __restrict__ hn, float* __restrict__ colsum, float* __restrict__ colsumsq,
    int N) {
  __shared__ alignas(16) h16 sWT1[64 * 128];  // nw1^T
  __shared__ alignas(16) h16 sWT2[64 * 64];   // nw2^T
  __shared__ float sB1[64], sB2[64];
  __shared__ alignas(16) h16 sF[4][16 * 32];
  __shared__ alignas(16) h16 sT1[4][16 * 64];

  const int tid = threadIdx.x;
  for (int i = tid; i < 64 * 128; i += 128) {
    const int n = i >> 7, k = i & 127;
    sWT1[i] = (h16)nw1[(size_t)k * 64 + n];
  }
  for (int i = tid; i < 64 * 64; i += 128) {
    const int n = i >> 6, k = i & 63;
    sWT2[i] = (h16)nw2[k * 64 + n];
  }
  if (tid < 64) { sB1[tid] = nb1[tid]; sB2[tid] = nb2[tid]; }
  __syncthreads();

  const int wave = tid >> 5, lane = tid & 31;
  const int n16 = lane & 15, hh = lane >> 4;
  h16* F = sF[wave]; h16* T1 = sT1[wave];

  const int waveGlobal = blockIdx.x * 4 + wave;
  const int totalWaves = gridDim.x * 4;
  const int nTiles = (N + 15) >> 4;

  for (int tile = waveGlobal; tile < nTiles; tile += totalWaves) {
    const int base = tile << 4;
    const int er = lane >> 1, coff = (lane & 1) * 16;
    const int row = base + er;
    const bool rv = row < N;
    const v8h vz = {};

    v8f acc[4] = {};
#pragma unroll
    for (int ck = 0; ck < 4; ++ck) {
      const h16* sp = (ck < 2) ? h : h_neighH;
      const int kb = (ck & 1) * 32 + coff;
      const v8h* p = (const v8h*)(sp + (size_t)(rv ? row : 0) * 64 + kb);
      const v8h q0 = p[0], q1 = p[1];
      *(v8h*)&F[er * 32 + coff]     = rv ? q0 : vz;
      *(v8h*)&F[er * 32 + coff + 8] = rv ? q1 : vz;
      const v16h a = load_a16(F, 32);
#pragma unroll
      for (int nt = 0; nt < 4; ++nt)
        acc[nt] = wmma16(a, load_b16(sWT1 + nt * 16 * 128, 128, ck * 32), acc[nt]);
    }
#pragma unroll
    for (int nt = 0; nt < 4; ++nt) {
      const int col = nt * 16 + n16;
      const float b = sB1[col];
#pragma unroll
      for (int v = 0; v < 8; ++v)
        T1[(v + 8 * hh) * 64 + col] = (h16)silu_f(acc[nt][v] + b);
    }

    v8f o[4] = {};
#pragma unroll
    for (int ck = 0; ck < 2; ++ck) {
      const v16h a = load_a16(T1 + ck * 32, 64);
#pragma unroll
      for (int nt = 0; nt < 4; ++nt)
        o[nt] = wmma16(a, load_b16(sWT2 + nt * 16 * 64, 64, ck * 32), o[nt]);
    }
#pragma unroll
    for (int nt = 0; nt < 4; ++nt) {
      const int col = nt * 16 + n16;
      const float b = sB2[col];
      float s8 = 0.f, q8 = 0.f;
#pragma unroll
      for (int v = 0; v < 8; ++v) {
        const int rw = base + v + 8 * hh;
        if (rw < N) {
          const float y = o[nt][v] + b;
          hn[(size_t)rw * 64 + col] = y;
          s8 += y; q8 += y * y;
        }
      }
      atomicAdd(&colsum[col], s8);
      atomicAdd(&colsumsq[col], q8);
    }
  }
}

__global__ void egnn_zero_kernel(float* __restrict__ p, long long n) {
  const long long i = (long long)blockIdx.x * blockDim.x + threadIdx.x;
  if (i < n) p[i] = 0.0f;
}

__global__ void egnn_cvt_kernel(const float* __restrict__ s, h16* __restrict__ d, long long n) {
  const long long i = (long long)blockIdx.x * blockDim.x + threadIdx.x;
  if (i < n) d[i] = (h16)s[i];
}

__global__ void egnn_bnstats_kernel(const float* __restrict__ colsum,
                                    const float* __restrict__ colsumsq,
                                    float* __restrict__ mu, float* __restrict__ inv, int N) {
  const int t = threadIdx.x;
  if (t < 64) {
    const float rn = 1.0f / (float)N;   // host-quality scalar divide, runs once
    const float m = colsum[t] * rn;
    const float v = colsumsq[t] * rn - m * m;   // biased variance
    mu[t] = m;
    inv[t] = rsqrtf(v + 1e-5f);
  }
}

__global__ void egnn_xupd_kernel(const float* __restrict__ xc, const float* __restrict__ xs,
                                 const float* __restrict__ degv, float* __restrict__ xn, int N) {
  const int i = blockIdx.x * blockDim.x + threadIdx.x;
  if (i < N) {
    const float rd = __builtin_amdgcn_rcpf(fmaxf(degv[i], 1.0f));
    xn[i * 3 + 0] = xc[i * 3 + 0] + xs[i * 3 + 0] * rd;
    xn[i * 3 + 1] = xc[i * 3 + 1] + xs[i * 3 + 1] * rd;
    xn[i * 3 + 2] = xc[i * 3 + 2] + xs[i * 3 + 2] * rd;
  }
}

__global__ void egnn_bnapply_kernel(const float* __restrict__ hn, const float* __restrict__ mu,
                                    const float* __restrict__ inv, const float* __restrict__ gamma,
                                    const float* __restrict__ beta, h16* __restrict__ hnextH,
                                    float* __restrict__ out, long long total, int first) {
  const long long i = (long long)blockIdx.x * blockDim.x + threadIdx.x;
  if (i < total) {
    const int c = (int)(i & 63);
    float y = (hn[i] - mu[c]) * inv[c] * gamma[c] + beta[c];
    y = fmaxf(y, 0.0f);
    hnextH[i] = (h16)y;
    out[i] = first ? y : (out[i] + y);
  }
}

extern "C" void kernel_launch(void* const* d_in, const int* in_sizes, int n_in,
                              void* d_out, int out_size, void* d_ws, size_t ws_size,
                              hipStream_t stream) {
  (void)n_in; (void)out_size; (void)ws_size;
  const float* node_feat = (const float*)d_in[0];
  const float* pos       = (const float*)d_in[1];
  const float* edge_feat = (const float*)d_in[2];
  const int*   src       = (const int*)d_in[3];
  const int*   dst       = (const int*)d_in[4];
  const float* ew1 = (const float*)d_in[5];
  const float* eb1 = (const float*)d_in[6];
  const float* ew2 = (const float*)d_in[7];
  const float* eb2 = (const float*)d_in[8];
  const float* cw1 = (const float*)d_in[9];
  const float* cb1 = (const float*)d_in[10];
  const float* cw2 = (const float*)d_in[11];
  const float* nw1 = (const float*)d_in[12];
  const float* nb1 = (const float*)d_in[13];
  const float* nw2 = (const float*)d_in[14];
  const float* nb2 = (const float*)d_in[15];
  const float* gamma = (const float*)d_in[16];
  const float* beta  = (const float*)d_in[17];

  const int N = in_sizes[0] / 64;
  const int E = in_sizes[3];
  const int L = in_sizes[6] / 64;

  float* w = (float*)d_ws;
  h16* hHA = (h16*)w;     w += (size_t)N * 32;   // f16 node features, ping
  h16* hHB = (h16*)w;     w += (size_t)N * 32;   // f16 node features, pong
  h16* efH = (h16*)w;     w += (size_t)E * 8;    // f16 edge features
  h16* hNeighH = (h16*)w; w += (size_t)N * 32;   // f16 aggregated messages
  float* xA = w;          w += (size_t)N * 3;
  float* xB = w;          w += (size_t)N * 3;
  float* h_neigh = w;     w += (size_t)N * 64;   // zero-region start
  float* x_sum = w;       w += (size_t)N * 3;
  float* degv = w;        w += (size_t)N;
  float* colsum = w;      w += 64;
  float* colsumsq = w;    w += 64;               // zero-region end
  float* hn = w;          w += (size_t)N * 64;
  float* bnMu = w;        w += 64;
  float* bnInv = w;

  const long long zn = (long long)N * 64 + (long long)N * 3 + N + 128;
  const long long tot = (long long)N * 64;
  const long long efn = (long long)E * 16;
  const int nTilesE = (E + 15) / 16;
  const int nTilesN = (N + 15) / 16;
  const int edgeBlocks = (nTilesE + 15) / 16;  // 4 waves/block, ~4 tiles/wave
  const int nodeBlocks = (nTilesN + 3) / 4;    // 4 waves/block, ~1 tile/wave

  // one-time f16 conversions (deterministic: redone every call)
  egnn_cvt_kernel<<<(int)((tot + 255) / 256), 256, 0, stream>>>(node_feat, hHA, tot);
  egnn_cvt_kernel<<<(int)((efn + 255) / 256), 256, 0, stream>>>(edge_feat, efH, efn);

  for (int l = 0; l < L; ++l) {
    const h16* h_cur = (l & 1) ? hHB : hHA;
    h16*       h_nx  = (l & 1) ? hHA : hHB;
    const float* x_cur = (l == 0) ? pos : ((l & 1) ? xA : xB);
    float*       x_nx  = (l & 1) ? xB : xA;

    egnn_zero_kernel<<<(int)((zn + 255) / 256), 256, 0, stream>>>(h_neigh, zn);
    egnn_edge_kernel<<<edgeBlocks, 128, 0, stream>>>(
        h_cur, x_cur, efH, src, dst,
        ew1 + (size_t)l * 145 * 64, eb1 + (size_t)l * 64,
        ew2 + (size_t)l * 64 * 64,  eb2 + (size_t)l * 64,
        cw1 + (size_t)l * 64 * 64,  cb1 + (size_t)l * 64,
        cw2 + (size_t)l * 64,
        h_neigh, x_sum, degv, E);
    egnn_cvt_kernel<<<(int)((tot + 255) / 256), 256, 0, stream>>>(h_neigh, hNeighH, tot);
    egnn_node_kernel<<<nodeBlocks, 128, 0, stream>>>(
        h_cur, hNeighH,
        nw1 + (size_t)l * 128 * 64, nb1 + (size_t)l * 64,
        nw2 + (size_t)l * 64 * 64,  nb2 + (size_t)l * 64,
        hn, colsum, colsumsq, N);
    egnn_bnstats_kernel<<<1, 64, 0, stream>>>(colsum, colsumsq, bnMu, bnInv, N);
    egnn_xupd_kernel<<<(N + 255) / 256, 256, 0, stream>>>(x_cur, x_sum, degv, x_nx, N);
    egnn_bnapply_kernel<<<(int)((tot + 255) / 256), 256, 0, stream>>>(
        hn, bnMu, bnInv, gamma + (size_t)l * 64, beta + (size_t)l * 64,
        h_nx, (float*)d_out, tot, l == 0);
  }
}
